// VITS_28793460752492
// MI455X (gfx1250) — matrix operand order repaired and checked by reading, not compiled, not run on hardware
//
#include <hip/hip_runtime.h>
#include <math.h>

#define B_   32
#define C_   192
#define TY_  1600
#define TX_  320
#define SEG_ 32
#define NEGV (-1e9f)

typedef __attribute__((ext_vector_type(16))) __bf16 v16bf;
typedef __attribute__((ext_vector_type(8)))  float  v8f;

static __device__ __forceinline__ unsigned short f2bf(float f) {
    // round-to-nearest-even f32 -> bf16
    unsigned int u = __float_as_uint(f);
    u += 0x7fffu + ((u >> 16) & 1u);
    return (unsigned short)(u >> 16);
}
static __device__ __forceinline__ float clip10(float v) {
    return fminf(10.f, fmaxf(-10.f, v));
}
static __device__ __forceinline__ float fix_num(float v) {
    // jnp.nan_to_num default: nan->0, +-inf -> +-FLT_MAX
    if (__builtin_isnan(v)) return 0.f;
    if (__builtin_isinf(v)) return v > 0.f ? 3.4028234663852886e38f
                                           : -3.4028234663852886e38f;
    return v;
}

// ---------------------------------------------------------------------------
// Kernel 0: per-(b,x) constant K = nc1 + nc4
//   nc1 = sum_c(-0.5*log(2pi) - lp),  nc4 = sum_c(-0.5*m^2*exp(-2 lp))
// ---------------------------------------------------------------------------
__global__ void k_const(const float* __restrict__ m_p,
                        const float* __restrict__ logs_p,
                        float* __restrict__ Kc) {
    int i = blockIdx.x * blockDim.x + threadIdx.x;
    if (i >= B_ * TX_) return;
    int b = i / TX_, x = i % TX_;
    float acc = -0.5f * 1.8378770664093453f * (float)C_;  // -C/2*log(2pi)
    for (int c = 0; c < C_; ++c) {
        int a = (b * C_ + c) * TX_ + x;
        float lp = clip10(logs_p[a]);
        float s  = expf(-2.f * lp);
        float m  = m_p[a];
        acc += -lp - 0.5f * m * m * s;
    }
    Kc[i] = acc;
}

// ---------------------------------------------------------------------------
// Kernel 1: neg_cent GEMM via v_wmma_f32_16x16x32_bf16.
//   Virtual A[y][k]: k=2c -> z_q[c,y], k=2c+1 -> -0.5*z_q[c,y]^2   (M=TY, K=384)
//   Virtual B[k][x]: k=2c -> m_p*s,    k=2c+1 -> s                 (N=TX)
// Block: 256 thr (8 waves) -> 64y x 64x output block.
// Each wave: 1 A-fragment reused across 2 B-fragments -> 2 WMMA per k-step.
// ---------------------------------------------------------------------------
__global__ void __launch_bounds__(256)
k_gemm(const float* __restrict__ z_q, const float* __restrict__ m_p,
       const float* __restrict__ logs_p, const float* __restrict__ Kc,
       float* __restrict__ neg) {
    __shared__ unsigned short As[64][34];  // [local y][local k], padded
    __shared__ unsigned short Bs[64][34];  // [local x][local k], padded

    const int tid   = threadIdx.x;
    const int b     = blockIdx.z;
    const int ybase = blockIdx.y * 64;
    const int xbase = blockIdx.x * 64;
    const int wave  = tid >> 5, lane = tid & 31;
    const int tyt   = wave & 3;            // y tile 0..3
    const int txp   = (wave >> 2) * 2;     // x tile pair base: 0 or 2
    const int mrow  = lane & 15, hi = lane >> 4;   // fragment lane decode

    v8f acc0 = {0.f, 0.f, 0.f, 0.f, 0.f, 0.f, 0.f, 0.f};
    v8f acc1 = {0.f, 0.f, 0.f, 0.f, 0.f, 0.f, 0.f, 0.f};

    for (int ks = 0; ks < 12; ++ks) {      // 12 * K32 = 384
        const int c0 = ks * 16;            // 16 channels -> 32 virtual K
        __syncthreads();
        // stage A: 64 y x 16 c (coalesced along y), emit z and -0.5 z^2
        #pragma unroll
        for (int i = 0; i < 4; ++i) {
            int idx = tid + i * 256;
            int cl = idx >> 6, yy = idx & 63;
            float z = z_q[(b * C_ + c0 + cl) * TY_ + ybase + yy];
            As[yy][2 * cl]     = f2bf(z);
            As[yy][2 * cl + 1] = f2bf(-0.5f * z * z);
        }
        // stage B: 64 x x 16 c, emit m_p*s and s (s can reach e^20 -> bf16 ok)
        #pragma unroll
        for (int i = 0; i < 4; ++i) {
            int idx = tid + i * 256;
            int cl = idx >> 6, xx = idx & 63;
            int a  = (b * C_ + c0 + cl) * TX_ + xbase + xx;
            float lp = clip10(logs_p[a]);
            float s  = expf(-2.f * lp);
            Bs[xx][2 * cl]     = f2bf(m_p[a] * s);
            Bs[xx][2 * cl + 1] = f2bf(s);
        }
        __syncthreads();

        // A fragment: 16-bit 16x32 layout (ISA 7.12.2):
        // lanes 0-15: K pairs {0..7,16..23}; lanes 16-31: +8
        union { v16bf v; unsigned short u[16]; } fa, fb0, fb1;
        #pragma unroll
        for (int p = 0; p < 8; ++p) {
            int kb = 2 * p + (p < 4 ? 0 : 8) + 8 * hi;
            fa.u[2 * p]     = As[tyt * 16 + mrow][kb];
            fa.u[2 * p + 1] = As[tyt * 16 + mrow][kb + 1];
        }
        // B fragments: lane%16 = N column, lanes 0-15 K {0..7,16..23}; lanes 16-31: +8
        #pragma unroll
        for (int e = 0; e < 16; ++e) {
            fb0.u[e] = Bs[txp * 16 + mrow][(e & 7) + 8 * hi + ((e >> 3) << 4)];
            fb1.u[e] = Bs[(txp + 1) * 16 + mrow][(e & 7) + 8 * hi + ((e >> 3) << 4)];
        }

        acc0 = __builtin_amdgcn_wmma_f32_16x16x32_bf16(
                   false, fa.v, false, fb0.v, (short)0, acc0, false, false);
        acc1 = __builtin_amdgcn_wmma_f32_16x16x32_bf16(
                   false, fa.v, false, fb1.v, (short)0, acc1, false, false);
    }

    // C/D layout: lane%16 = N, vgpr e -> M = e + 8*(lane/16)
    const int x0 = xbase + txp * 16 + mrow;
    const int x1 = x0 + 16;
    const float k0 = Kc[b * TX_ + x0];
    const float k1 = Kc[b * TX_ + x1];
    #pragma unroll
    for (int e = 0; e < 8; ++e) {
        int y = ybase + tyt * 16 + e + 8 * hi;
        neg[(b * TY_ + y) * TX_ + x0] = acc0[e] + k0;
        neg[(b * TY_ + y) * TX_ + x1] = acc1[e] + k1;
    }
}

// ---------------------------------------------------------------------------
// Kernel 2: forward DP.  One WG per batch, 320 threads (10 wave32).
// Q rows ping-pong in LDS; direction bits packed via wave ballot.
// Latency-bound serial loop: prefetch next row (L2-resident) behind barrier.
// ---------------------------------------------------------------------------
__global__ void __launch_bounds__(320)
k_dp(const float* __restrict__ neg, const int* __restrict__ xls,
     const int* __restrict__ yls, unsigned int* __restrict__ dirs) {
    __shared__ float Q[2][TX_];
    const int x  = threadIdx.x;
    const int b  = blockIdx.x;
    const int xl = xls[b], yl = yls[b];
    const int lane = x & 31, wv = x >> 5;

    float v0 = NEGV;
    if (x == 0 && xl > 0 && yl > 0) v0 = neg[(b * TY_) * TX_];
    Q[0][x] = (x == 0) ? v0 : NEGV;
    __syncthreads();

    int cur = 0;
    for (int y = 1; y < TY_; ++y) {
        float v = neg[(b * TY_ + y) * TX_ + x];
        if (y + 1 < TY_)  // hide next row's L2 latency behind this iteration
            __builtin_prefetch(&neg[(b * TY_ + y + 1) * TX_ + x], 0, 3);
        if (x >= xl || y >= yl) v = NEGV;
        float qp = Q[cur][x];
        float sh = (x > 0) ? Q[cur][x - 1] : NEGV;
        bool  take = sh > qp;                       // strict, matches cython
        unsigned int mask = (unsigned int)__ballot(take ? 1 : 0);
        if (lane == 0) dirs[(b * TY_ + y) * 10 + wv] = mask;
        Q[cur ^ 1][x] = v + fmaxf(qp, sh);
        __syncthreads();
        cur ^= 1;
    }
}

// ---------------------------------------------------------------------------
// Kernel 3: backtrack (thread 0 per batch) + w counts + logw_ output.
// ---------------------------------------------------------------------------
__global__ void __launch_bounds__(320)
k_backtrack(const unsigned int* __restrict__ dirs, const int* __restrict__ xls,
            const int* __restrict__ yls, int* __restrict__ path,
            float* __restrict__ logw) {
    __shared__ float w[TX_];
    const int b  = blockIdx.x;
    const int xl = xls[b], yl = yls[b];
    w[threadIdx.x] = 0.f;
    __syncthreads();
    if (threadIdx.x == 0) {
        int x = xl - 1;
        for (int y = TY_ - 1; y >= 0; --y) {
            if (y < yl) {
                path[b * TY_ + y] = x;
                w[x] += 1.f;
                if (y > 0) {
                    unsigned int word = dirs[(b * TY_ + y) * 10 + (x >> 5)];
                    x -= (int)((word >> (x & 31)) & 1u);
                }
            } else {
                path[b * TY_ + y] = -1;
            }
        }
    }
    __syncthreads();
    float lw = logf(w[threadIdx.x] + 1e-6f);
    logw[b * TX_ + threadIdx.x] = (threadIdx.x < xl) ? lw : 0.f;
}

// ---------------------------------------------------------------------------
// Kernel 4: materialize attn [B,TY,TX] (full write; d_out is poisoned).
// ---------------------------------------------------------------------------
__global__ void __launch_bounds__(320)
k_attn(const int* __restrict__ path, float* __restrict__ attn) {
    __shared__ int px;
    const int y = blockIdx.x, b = blockIdx.y;
    if (threadIdx.x == 0) px = path[b * TY_ + y];
    __syncthreads();
    attn[(b * TY_ + y) * TX_ + threadIdx.x] = (threadIdx.x == px) ? 1.f : 0.f;
}

// ---------------------------------------------------------------------------
// Kernel 5: stacked [6,B,C,SEG] — hard-path einsums are gathers; only the
// 32 sliced frames are needed.
// ---------------------------------------------------------------------------
__global__ void k_gather(const float* __restrict__ zp, const float* __restrict__ mp,
                         const float* __restrict__ lsp, const float* __restrict__ zq,
                         const float* __restrict__ mq, const float* __restrict__ lsq,
                         const int* __restrict__ ids, const int* __restrict__ path,
                         float* __restrict__ out) {
    int idx = blockIdx.x * blockDim.x + threadIdx.x;
    const int N = 6 * B_ * C_ * SEG_;
    if (idx >= N) return;
    int t = idx % SEG_;
    int c = (idx / SEG_) % C_;
    int b = (idx / (SEG_ * C_)) % B_;
    int s = idx / (SEG_ * C_ * B_);
    int y = ids[b] + t;
    float v;
    if (s == 1)      v = zq[(b * C_ + c) * TY_ + y];
    else if (s == 4) v = mq[(b * C_ + c) * TY_ + y];
    else if (s == 5) v = lsq[(b * C_ + c) * TY_ + y];
    else {
        int px = path[b * TY_ + y];
        if (px < 0) {
            v = 0.f;                         // zero attn row -> zero gather
        } else {
            int a = (b * C_ + c) * TX_ + px;
            if (s == 0)      v = fix_num(zp[a]);
            else if (s == 2) v = fix_num(mp[a]);
            else             v = clip10(lsp[a]);
        }
    }
    out[idx] = v;
}

// ---------------------------------------------------------------------------
extern "C" void kernel_launch(void* const* d_in, const int* in_sizes, int n_in,
                              void* d_out, int out_size, void* d_ws, size_t ws_size,
                              hipStream_t stream) {
    const float* z_p    = (const float*)d_in[0];
    const float* m_p    = (const float*)d_in[1];
    const float* logs_p = (const float*)d_in[2];
    const float* z_q    = (const float*)d_in[3];
    const float* m_q    = (const float*)d_in[4];
    const float* logs_q = (const float*)d_in[5];
    const int*   xl     = (const int*)d_in[6];
    const int*   yl     = (const int*)d_in[7];
    const int*   ids    = (const int*)d_in[8];
    (void)in_sizes; (void)n_in; (void)out_size; (void)ws_size;

    // workspace layout (bytes)
    char* ws = (char*)d_ws;
    float*        neg  = (float*)ws;                          // 32*1600*320*4 = 65,536,000
    float*        Kc   = (float*)(ws + 65536000);             // 32*320*4      =     40,960
    unsigned int* dirs = (unsigned int*)(ws + 65576960);      // 32*1600*10*4  =  2,048,000
    int*          path = (int*)(ws + 67624960);               // 32*1600*4     =    204,800

    float* out_stacked = (float*)d_out;                       // 6*32*192*32
    float* out_logw    = out_stacked + 6 * B_ * C_ * SEG_;    // 32*320
    float* out_attn    = out_logw + B_ * TX_;                 // 32*1600*320

    k_const<<<dim3((B_ * TX_ + 255) / 256), 256, 0, stream>>>(m_p, logs_p, Kc);
    k_gemm<<<dim3(TX_ / 64, TY_ / 64, B_), 256, 0, stream>>>(z_q, m_p, logs_p, Kc, neg);
    k_dp<<<dim3(B_), 320, 0, stream>>>(neg, xl, yl, dirs);
    k_backtrack<<<dim3(B_), 320, 0, stream>>>(dirs, xl, yl, path, out_logw);
    k_attn<<<dim3(TY_, B_), 320, 0, stream>>>(path, out_attn);
    k_gather<<<dim3((6 * B_ * C_ * SEG_ + 255) / 256), 256, 0, stream>>>(
        z_p, m_p, logs_p, z_q, m_q, logs_q, ids, path, out_stacked);
}
